// LayerNormLSTMCell_12128987644431
// MI455X (gfx1250) — matrix operand order repaired and checked by reading, not compiled, not run on hardware
//
#include <hip/hip_runtime.h>
#include <cstdint>

// ---------------------------------------------------------------------------
// LayerNorm LSTM cell for MI455X (gfx1250, wave32, WMMA, async-to-LDS)
//   Phase 0: convert x/hx/W_ih/W_hh f32 -> bf16 (v_perm packing)
//   Phase 1: two bf16-WMMA GEMMs, tiles staged via GLOBAL_LOAD_ASYNC_TO_LDS
//   Phase 2: fused epilogue: LN(IG)+LN(HG) -> gates -> LN(c) -> hy/cy
// ---------------------------------------------------------------------------

typedef __attribute__((ext_vector_type(16))) __bf16 v16bf;
typedef __attribute__((ext_vector_type(8)))  float  v8f;
typedef __attribute__((ext_vector_type(4)))  unsigned int u32x4;
typedef __attribute__((ext_vector_type(4)))  int v4i;

#define BDIM 8192
#define KDIM 1024
#define NDIM 4096
#define KT   32            // K per LDS stage (one WMMA worth)
#define NKT  (KDIM / KT)   // 32
#define LDST 20            // dwords per 32-bf16 LDS row (64B data + 16B pad = 80B)

// ---- CDNA5 async-copy plumbing (guarded: falls back to plain copies) ------
#if __has_builtin(__builtin_amdgcn_global_load_async_to_lds_b128)
#define HAS_ASYNC_LDS 1
typedef __attribute__((address_space(1))) v4i* gp128;  // int4 __device__*
typedef __attribute__((address_space(3))) v4i* lp128;  // int4 __shared__*
__device__ inline gp128 gcast(const void* p) { return (gp128)(uintptr_t)p; }
__device__ inline lp128 lcast(void* p)       { return (lp128)(unsigned)(uintptr_t)p; }
#endif

#if __has_builtin(__builtin_amdgcn_s_wait_asynccnt)
#define WAIT_ASYNC(n) __builtin_amdgcn_s_wait_asynccnt(n)
#elif defined(HAS_ASYNC_LDS)
#define WAIT_ASYNC(n) asm volatile("s_wait_asynccnt " #n ::: "memory")
#else
#define WAIT_ASYNC(n)
#endif

// two f32 -> packed bf16x2, round-half-up then byte-select via v_perm_b32
__device__ inline unsigned pk2(float a, float b) {
    unsigned ua = __float_as_uint(a) + 0x8000u;
    unsigned ub = __float_as_uint(b) + 0x8000u;
    return __builtin_amdgcn_perm(ub, ua, 0x07060302u);  // [b.hi16 | a.hi16]
}

// ---------------------------------------------------------------------------
// Phase 0: bulk f32 -> bf16 conversion (8 elems / thread)
// ---------------------------------------------------------------------------
__global__ void __launch_bounds__(256)
cvt_f32_bf16(const float* __restrict__ src, unsigned* __restrict__ dst, int n8) {
    int i = blockIdx.x * 256 + threadIdx.x;
    if (i >= n8) return;
    const float4* s = (const float4*)src + (size_t)i * 2;
    float4 a = s[0], b = s[1];
    u32x4 v;
    v.x = pk2(a.x, a.y); v.y = pk2(a.z, a.w);
    v.z = pk2(b.x, b.y); v.w = pk2(b.z, b.w);
    *(u32x4*)(dst + (size_t)i * 4) = v;
}

// two 16B LDS reads -> one 16-lane bf16 fragment
__device__ inline v16bf ldfrag(const unsigned* p0, const unsigned* p1) {
    union { u32x4 q[2]; v16bf v; } u;
    u.q[0] = *(const u32x4*)p0;
    u.q[1] = *(const u32x4*)p1;
    return u.v;
}

// ---------------------------------------------------------------------------
// Phase 1 GEMM: C[M,N] = A[M,K](bf16) * W[N,K](bf16)^T, f32 out
// block tile 128x128, 8 waves, wave tile 64x32, async double-buffered LDS
// ---------------------------------------------------------------------------
__global__ void __launch_bounds__(256, 2)
gemm_bf16_wmma(const __bf16* __restrict__ A, const __bf16* __restrict__ W,
               float* __restrict__ C) {
    __shared__ unsigned Abuf[2][128 * LDST];
    __shared__ unsigned Bbuf[2][128 * LDST];

    const int t    = threadIdx.x;
    const int lane = t & 31;
    const int wid  = t >> 5;
    const int wm   = (wid & 1) * 64;
    const int wn   = (wid >> 1) * 32;
    const int lm   = lane & 15;
    const int hf   = lane >> 4;

    const int gn0 = blockIdx.x * 128;
    const int gm0 = blockIdx.y * 128;

    v8f acc[4][2] = {};

    // stage one K-tile into LDS buffer `buf` (4x b128 per thread: 2 A + 2 B)
    auto issue = [&](int kt, int buf) {
        const int k0 = kt * KT;
#pragma unroll
        for (int i = 0; i < 2; ++i) {
            int idx = i * 256 + t;
            int row = idx >> 2, c = idx & 3;          // 4x 16B chunks per 64B row
            const void* ga = A + (size_t)(gm0 + row) * KDIM + k0 + c * 8;
            const void* gw = W + (size_t)(gn0 + row) * KDIM + k0 + c * 8;
            void* la = (char*)&Abuf[buf][row * LDST] + c * 16;
            void* lw = (char*)&Bbuf[buf][row * LDST] + c * 16;
#ifdef HAS_ASYNC_LDS
            __builtin_amdgcn_global_load_async_to_lds_b128(gcast(ga), lcast(la), 0, 0);
            __builtin_amdgcn_global_load_async_to_lds_b128(gcast(gw), lcast(lw), 0, 0);
#else
            *(u32x4*)la = *(const u32x4*)ga;
            *(u32x4*)lw = *(const u32x4*)gw;
#endif
        }
    };

    issue(0, 0);

    for (int kt = 0; kt < NKT; ++kt) {
        const int cur = kt & 1;

        if (kt + 1 < NKT) {
            issue(kt + 1, cur ^ 1);
            WAIT_ASYNC(4);              // retire this tile's 4 asyncs, keep next 4 in flight
        } else {
            WAIT_ASYNC(0);
        }
        if (kt + 2 < NKT) {             // warm L2 for the tile after next
            const int k2 = (kt + 2) * KT;
            int row = t >> 2, c = t & 3;
            __builtin_prefetch(A + (size_t)(gm0 + row) * KDIM + k2 + c * 8, 0, 1);
            __builtin_prefetch(W + (size_t)(gn0 + row) * KDIM + k2 + c * 8, 0, 1);
        }
        __syncthreads();

        const unsigned* As = Abuf[cur];
        const unsigned* Bs = Bbuf[cur];
        v16bf af[4], bf[2];
#pragma unroll
        for (int mi = 0; mi < 4; ++mi) {
            // A 16x32 bf16: lane half -> K 0..7 / 8..15, second read +16
            const unsigned* p = As + (wm + mi * 16 + lm) * LDST + hf * 4;
            af[mi] = ldfrag(p, p + 8);
        }
#pragma unroll
        for (int ni = 0; ni < 2; ++ni) {
            // B 32x16 bf16: lane half -> K 0..7 / 8..15, second read +16
            const unsigned* p = Bs + (wn + ni * 16 + lm) * LDST + hf * 4;
            bf[ni] = ldfrag(p, p + 8);
        }
#pragma unroll
        for (int mi = 0; mi < 4; ++mi)
#pragma unroll
            for (int ni = 0; ni < 2; ++ni)
                acc[mi][ni] = __builtin_amdgcn_wmma_f32_16x16x32_bf16(
                    false, af[mi], false, bf[ni], (short)0, acc[mi][ni],
                    false, false);

        __syncthreads();                // all waves done with `cur` before it is re-filled
    }

    // C store: 16x16 f32 layout (vgpr r -> M = r + 8*half, N = lane%16)
#pragma unroll
    for (int mi = 0; mi < 4; ++mi)
#pragma unroll
        for (int ni = 0; ni < 2; ++ni) {
            const int col = gn0 + wn + ni * 16 + lm;
#pragma unroll
            for (int r = 0; r < 8; ++r) {
                const int row = gm0 + wm + mi * 16 + r + hf * 8;
                C[(size_t)row * NDIM + col] = acc[mi][ni][r];
            }
        }
}

// ---------------------------------------------------------------------------
// Phase 2: fused epilogue, one wave32 per batch row
// ---------------------------------------------------------------------------
__device__ inline float wave_sum(float v) {
#pragma unroll
    for (int off = 16; off > 0; off >>= 1) v += __shfl_xor(v, off, 32);
    return v;
}
__device__ inline void sig4(float4& v) {
    float* p = &v.x;
#pragma unroll
    for (int c = 0; c < 4; ++c) p[c] = 1.f / (1.f + __expf(-p[c]));
}
__device__ inline void tanh4(float4& v) {
    float* p = &v.x;
#pragma unroll
    for (int c = 0; c < 4; ++c) {
        float e = __expf(-2.f * p[c]);
        p[c] = (1.f - e) / (1.f + e);
    }
}
__device__ inline float4 gate4(const float* ig, const float* hg,
                               const float* gi, const float* bi,
                               const float* gh, const float* bh, int j,
                               float mui, float rsi, float muh, float rsh) {
    float4 a = *(const float4*)(ig + j), b = *(const float4*)(hg + j);
    float4 Gi = *(const float4*)(gi + j), Bi = *(const float4*)(bi + j);
    float4 Gh = *(const float4*)(gh + j), Bh = *(const float4*)(bh + j);
    float4 r;
    r.x = Gi.x * (a.x - mui) * rsi + Bi.x + Gh.x * (b.x - muh) * rsh + Bh.x;
    r.y = Gi.y * (a.y - mui) * rsi + Bi.y + Gh.y * (b.y - muh) * rsh + Bh.y;
    r.z = Gi.z * (a.z - mui) * rsi + Bi.z + Gh.z * (b.z - muh) * rsh + Bh.z;
    r.w = Gi.w * (a.w - mui) * rsi + Bi.w + Gh.w * (b.w - muh) * rsh + Bh.w;
    return r;
}

__global__ void __launch_bounds__(256)
lstm_epilogue(const float* __restrict__ IG, const float* __restrict__ HG,
              const float* __restrict__ cx,
              const float* __restrict__ gi, const float* __restrict__ bi,
              const float* __restrict__ gh, const float* __restrict__ bh,
              const float* __restrict__ gc, const float* __restrict__ bc,
              float* __restrict__ out) {
    const int t = threadIdx.x, lane = t & 31, wid = t >> 5;
    const int row = blockIdx.x * 8 + wid;
    const float* ig  = IG + (size_t)row * NDIM;
    const float* hg  = HG + (size_t)row * NDIM;
    const float* cxr = cx + (size_t)row * 1024;

    float si = 0.f, si2 = 0.f, sh = 0.f, sh2 = 0.f;
#pragma unroll 4
    for (int j4 = 0; j4 < 32; ++j4) {
        int j = (j4 * 32 + lane) * 4;
        float4 a = *(const float4*)(ig + j);
        float4 b = *(const float4*)(hg + j);
        si  += a.x + a.y + a.z + a.w;
        si2 += a.x * a.x + a.y * a.y + a.z * a.z + a.w * a.w;
        sh  += b.x + b.y + b.z + b.w;
        sh2 += b.x * b.x + b.y * b.y + b.z * b.z + b.w * b.w;
    }
    si = wave_sum(si); si2 = wave_sum(si2);
    sh = wave_sum(sh); sh2 = wave_sum(sh2);
    const float inv4h = 1.f / 4096.f;
    const float mui = si * inv4h, muh = sh * inv4h;
    const float rsi = rsqrtf(si2 * inv4h - mui * mui + 1e-5f);
    const float rsh = rsqrtf(sh2 * inv4h - muh * muh + 1e-5f);

    float4 cp[8], oo[8];
    float sc = 0.f, sc2 = 0.f;
#pragma unroll
    for (int it = 0; it < 8; ++it) {
        int h = (it * 32 + lane) * 4;
        float4 vi = gate4(ig, hg, gi, bi, gh, bh, h,        mui, rsi, muh, rsh);
        float4 vf = gate4(ig, hg, gi, bi, gh, bh, h + 1024, mui, rsi, muh, rsh);
        float4 vg = gate4(ig, hg, gi, bi, gh, bh, h + 2048, mui, rsi, muh, rsh);
        float4 vo = gate4(ig, hg, gi, bi, gh, bh, h + 3072, mui, rsi, muh, rsh);
        sig4(vi); sig4(vf); sig4(vo); tanh4(vg);
        float4 c4 = *(const float4*)(cxr + h);
        float4 c;
        c.x = vf.x * c4.x + vi.x * vg.x;
        c.y = vf.y * c4.y + vi.y * vg.y;
        c.z = vf.z * c4.z + vi.z * vg.z;
        c.w = vf.w * c4.w + vi.w * vg.w;
        cp[it] = c; oo[it] = vo;
        sc  += c.x + c.y + c.z + c.w;
        sc2 += c.x * c.x + c.y * c.y + c.z * c.z + c.w * c.w;
    }
    sc = wave_sum(sc); sc2 = wave_sum(sc2);
    const float inv_h = 1.f / 1024.f;
    const float muc = sc * inv_h;
    const float rsc = rsqrtf(sc2 * inv_h - muc * muc + 1e-5f);

    const size_t OFS = (size_t)BDIM * 1024;
#pragma unroll
    for (int it = 0; it < 8; ++it) {
        int h = (it * 32 + lane) * 4;
        float4 Gc = *(const float4*)(gc + h), Bc = *(const float4*)(bc + h);
        float4 cy, th;
        cy.x = (cp[it].x - muc) * rsc * Gc.x + Bc.x;
        cy.y = (cp[it].y - muc) * rsc * Gc.y + Bc.y;
        cy.z = (cp[it].z - muc) * rsc * Gc.z + Bc.z;
        cy.w = (cp[it].w - muc) * rsc * Gc.w + Bc.w;
        th = cy; tanh4(th);
        float4 hy;
        hy.x = oo[it].x * th.x; hy.y = oo[it].y * th.y;
        hy.z = oo[it].z * th.z; hy.w = oo[it].w * th.w;
        size_t o = (size_t)row * 1024 + h;
        *(float4*)(out + o)           = hy;
        *(float4*)(out + OFS + o)     = hy;
        *(float4*)(out + 2 * OFS + o) = cy;
    }
}

// ---------------------------------------------------------------------------
extern "C" void kernel_launch(void* const* d_in, const int* in_sizes, int n_in,
                              void* d_out, int out_size, void* d_ws, size_t ws_size,
                              hipStream_t stream) {
    const float* x    = (const float*)d_in[0];
    const float* hx   = (const float*)d_in[1];
    const float* cx   = (const float*)d_in[2];
    const float* W_ih = (const float*)d_in[3];
    const float* W_hh = (const float*)d_in[4];
    const float* gi   = (const float*)d_in[5];
    const float* bi   = (const float*)d_in[6];
    const float* gh   = (const float*)d_in[7];
    const float* bh   = (const float*)d_in[8];
    const float* gc   = (const float*)d_in[9];
    const float* bc   = (const float*)d_in[10];
    float* out = (float*)d_out;

    // workspace layout
    float*  IG   = (float*)d_ws;                          // 128 MB
    float*  HG   = IG + (size_t)BDIM * NDIM;              // 128 MB
    __bf16* Xbf  = (__bf16*)(HG + (size_t)BDIM * NDIM);   // 16 MB
    __bf16* Hbf  = Xbf + (size_t)BDIM * KDIM;             // 16 MB
    __bf16* Wibf = Hbf + (size_t)BDIM * KDIM;             // 8 MB
    __bf16* Whbf = Wibf + (size_t)NDIM * KDIM;            // 8 MB

    // Phase 0: f32 -> bf16
    const int nx8 = BDIM * KDIM / 8, nw8 = NDIM * KDIM / 8;
    cvt_f32_bf16<<<dim3(nx8 / 256), dim3(256), 0, stream>>>(x,    (unsigned*)Xbf,  nx8);
    cvt_f32_bf16<<<dim3(nx8 / 256), dim3(256), 0, stream>>>(hx,   (unsigned*)Hbf,  nx8);
    cvt_f32_bf16<<<dim3(nw8 / 256), dim3(256), 0, stream>>>(W_ih, (unsigned*)Wibf, nw8);
    cvt_f32_bf16<<<dim3(nw8 / 256), dim3(256), 0, stream>>>(W_hh, (unsigned*)Whbf, nw8);

    // Phase 1: WMMA GEMMs
    dim3 gblk(256), ggrd(NDIM / 128, BDIM / 128);
    gemm_bf16_wmma<<<ggrd, gblk, 0, stream>>>(Xbf, Wibf, IG);
    gemm_bf16_wmma<<<ggrd, gblk, 0, stream>>>(Hbf, Whbf, HG);

    // Phase 2: fused layernorm/LSTM epilogue
    lstm_epilogue<<<dim3(BDIM / 8), dim3(256), 0, stream>>>(
        IG, HG, cx, gi, bi, gh, bh, gc, bc, out);
}
